// NodeDecoder_71760313582002
// MI455X (gfx1250) — matrix operand order repaired and checked by reading, not compiled, not run on hardware
//
#include <hip/hip_runtime.h>

#define TM   128      // rows per workgroup
#define DD   256      // node input dim
#define MD   64       // metric embedding dim
#define K1   320      // DD + MD
#define H1   256      // hidden
#define H2   128      // hidden/2
#define TT   16384    // total metrics per batch

#define W1T_ELEMS (H1 * K1)   // 256*320 bf16, [N][K]
#define W2T_ELEMS (H2 * H1)   // 128*256 bf16, [N][K]

typedef __attribute__((ext_vector_type(16))) __bf16 v16bf;
typedef __attribute__((ext_vector_type(8)))  float  v8f;
typedef __attribute__((ext_vector_type(4)))  float  f32x4;   // clang vector (nontemporal-ok)

__device__ __forceinline__ unsigned short f2bf(float f) {
  union { __bf16 h; unsigned short s; } u;
  u.h = (__bf16)f;              // hardware f32->bf16 cvt (RNE)
  return u.s;
}

__device__ __forceinline__ v8f wmma_bf16(v16bf a, v16bf b, v8f c) {
  return __builtin_amdgcn_wmma_f32_16x16x32_bf16(false, a, false, b, (short)0, c,
                                                 false, false);
}

// A fragment: 16x32 bf16 tile from LDS, row-major with `stride` elements.
// Lane L holds row M = L%16; K runs {16h..16h+15}, h = L/16.
__device__ __forceinline__ v16bf load_afrag(const unsigned short* s, int stride,
                                            int m0, int k0, int lane) {
  const unsigned short* p = s + (m0 + (lane & 15)) * stride + k0 + 16 * (lane >> 4);
  union { uint4 q[2]; v16bf v; } u;
  u.q[0] = *(const uint4*)(p);        // ds_load_b128
  u.q[1] = *(const uint4*)(p + 8);    // ds_load_b128
  return u.v;
}

// B fragment: 32x16 bf16 tile from pre-transposed bf16 weights Wt[n][k].
// Lane L holds column N = n0 + L%16; elements e -> K = k0 + 16*(L/16) + e,
// which is 32 contiguous bytes -> two global_load_b128.
__device__ __forceinline__ v16bf load_wfrag(const unsigned short* __restrict__ Wt,
                                            int ldk, int k0, int n0, int lane) {
  const unsigned short* p = Wt + (size_t)(n0 + (lane & 15)) * ldk + k0 + 16 * (lane >> 4);
  union { uint4 q[2]; v16bf v; } u;
  u.q[0] = *(const uint4*)(p);
  u.q[1] = *(const uint4*)(p + 8);
  return u.v;
}

// ---- one-time weight prep: fp32 [K][N] -> bf16 [N][K] in workspace ----
__global__ __launch_bounds__(256) void prep_weights(
    const float* __restrict__ W1, const float* __restrict__ W2,
    const float* __restrict__ W3, unsigned short* __restrict__ ws) {
  unsigned short* W1t = ws;
  unsigned short* W2t = ws + W1T_ELEMS;
  unsigned short* W3b = W2t + W2T_ELEMS;
  const int idx = blockIdx.x * 256 + threadIdx.x;
  const int stride = gridDim.x * 256;
  for (int i = idx; i < W1T_ELEMS; i += stride) {
    const int n = i / K1, k = i - n * K1;
    W1t[i] = f2bf(W1[k * H1 + n]);
  }
  for (int i = idx; i < W2T_ELEMS; i += stride) {
    const int n = i / H1, k = i - n * H1;
    W2t[i] = f2bf(W2[k * H2 + n]);
  }
  if (idx < H2) W3b[idx] = f2bf(W3[idx]);
}

__global__ __launch_bounds__(256) void node_decoder_mlp(
    const float* __restrict__ hinst,   // [B, 512, 256]
    const float* __restrict__ memb,    // [B, TT, 64]
    const int*   __restrict__ m2i,     // [TT]
    const unsigned short* __restrict__ ws,   // W1t | W2t | W3b (bf16)
    const float* __restrict__ b1, const float* __restrict__ b2,
    const float* __restrict__ b3,
    float* __restrict__ out)           // [B*TT]
{
  extern __shared__ unsigned short smem[];
  unsigned short* xs  = smem;            // TM*K1 bf16 (80 KB)
  unsigned short* h1s = smem + TM * K1;  // TM*H1 bf16 (64 KB)
  unsigned short* h2s = smem;            // TM*H2 bf16 (aliases dead xs region)

  const unsigned short* W1t = ws;
  const unsigned short* W2t = ws + W1T_ELEMS;
  const unsigned short* W3b = W2t + W2T_ELEMS;

  const int tid  = threadIdx.x;
  const int wave = tid >> 5;
  const int lane = tid & 31;
  const int wg   = blockIdx.x;
  const int bb   = (wg * TM) / TT;       // batch index (TM divides TT)
  const int t0   = (wg * TM) % TT;       // first metric row of this tile

  // Warm bf16 weight cachelines (global_prefetch_b8).
  for (int off = tid * 32; off < W1T_ELEMS; off += 256 * 32) __builtin_prefetch(W1t + off, 0, 3);
  for (int off = tid * 32; off < W2T_ELEMS; off += 256 * 32) __builtin_prefetch(W2t + off, 0, 3);

  // ---- stage x = [gather(h_inst), metric_emb] as bf16 into LDS ----
  {
    const int rg  = tid >> 6;   // 0..3 : 4 rows in flight
    const int c64 = tid & 63;   // 64 lanes per row, float4 each covers 256
    for (int i = rg; i < TM; i += 4) {
      const int t    = t0 + i;
      const int inst = m2i[t];
      const f32x4 v = *(const f32x4*)(hinst + ((size_t)bb * 512 + inst) * DD + c64 * 4);
      union { unsigned short s[4]; uint2 u; } p;
      p.s[0] = f2bf(v.x); p.s[1] = f2bf(v.y); p.s[2] = f2bf(v.z); p.s[3] = f2bf(v.w);
      *(uint2*)(xs + i * K1 + c64 * 4) = p.u;
      if (c64 < 16) {  // 16 lanes x float4 cover the 64 metric features
        const f32x4* mp = (const f32x4*)(memb + ((size_t)bb * TT + t) * MD + c64 * 4);
        const f32x4 m = __builtin_nontemporal_load(mp);   // streamed once; keep L2 for gather
        union { unsigned short s[4]; uint2 u; } q;
        q.s[0] = f2bf(m.x); q.s[1] = f2bf(m.y); q.s[2] = f2bf(m.z); q.s[3] = f2bf(m.w);
        *(uint2*)(xs + i * K1 + DD + c64 * 4) = q.u;
      }
    }
  }
  __syncthreads();

  // ---- layer 1: h1 = relu(x @ W1 + b1)   [128x320]@[320x256] ----
  const int nt0 = wave * 2, nt1 = wave * 2 + 1;  // this wave's two N-tiles
  v8f acc1[8][2];
  {
    const float bv0 = b1[nt0 * 16 + (lane & 15)];
    const float bv1 = b1[nt1 * 16 + (lane & 15)];
    #pragma unroll
    for (int m = 0; m < 8; ++m)
      #pragma unroll
      for (int i = 0; i < 8; ++i) { acc1[m][0][i] = bv0; acc1[m][1][i] = bv1; }
  }
  for (int ks = 0; ks < K1 / 32; ++ks) {
    const int k0 = ks * 32;
    const v16bf B0 = load_wfrag(W1t, K1, k0, nt0 * 16, lane);
    const v16bf B1 = load_wfrag(W1t, K1, k0, nt1 * 16, lane);
    #pragma unroll
    for (int m = 0; m < 8; ++m) {
      const v16bf A = load_afrag(xs, K1, m * 16, k0, lane);
      acc1[m][0] = wmma_bf16(A, B0, acc1[m][0]);
      acc1[m][1] = wmma_bf16(A, B1, acc1[m][1]);
    }
  }
  // relu -> h1s (D layout: lane L -> N=L%16; VGPR i -> M = i + 8*(L/16))
  #pragma unroll
  for (int j = 0; j < 2; ++j) {
    const int nb = (nt0 + j) * 16 + (lane & 15);
    #pragma unroll
    for (int m = 0; m < 8; ++m)
      #pragma unroll
      for (int i = 0; i < 8; ++i)
        h1s[(m * 16 + i + 8 * (lane >> 4)) * H1 + nb] = f2bf(fmaxf(acc1[m][j][i], 0.f));
  }
  __syncthreads();

  // ---- layer 2: h2 = relu(h1 @ W2 + b2)   [128x256]@[256x128] ----
  v8f acc2[8];
  {
    const float bv = b2[wave * 16 + (lane & 15)];
    #pragma unroll
    for (int m = 0; m < 8; ++m)
      #pragma unroll
      for (int i = 0; i < 8; ++i) acc2[m][i] = bv;
  }
  for (int ks = 0; ks < H1 / 32; ++ks) {
    const int k0 = ks * 32;
    const v16bf Bf = load_wfrag(W2t, H1, k0, wave * 16, lane);
    #pragma unroll
    for (int m = 0; m < 8; ++m) {
      const v16bf A = load_afrag(h1s, H1, m * 16, k0, lane);
      acc2[m] = wmma_bf16(A, Bf, acc2[m]);
    }
  }
  {
    const int nb = wave * 16 + (lane & 15);
    #pragma unroll
    for (int m = 0; m < 8; ++m)
      #pragma unroll
      for (int i = 0; i < 8; ++i)
        h2s[(m * 16 + i + 8 * (lane >> 4)) * H2 + nb] = f2bf(fmaxf(acc2[m][i], 0.f));
  }
  __syncthreads();

  // ---- layer 3: out = h2 @ W3 + b3  (W3 padded into a 16-wide N-tile) ----
  v8f acc3;
  {
    const float bv = b3[0];
    #pragma unroll
    for (int i = 0; i < 8; ++i) acc3[i] = bv;
  }
  #pragma unroll
  for (int ks = 0; ks < H2 / 32; ++ks) {
    const int k0 = ks * 32;
    union { uint4 q[2]; unsigned short s[16]; v16bf v; } u;
    u.q[0] = make_uint4(0, 0, 0, 0);
    u.q[1] = make_uint4(0, 0, 0, 0);
    if ((lane & 15) == 0) {           // only column N=0 carries W3
      const unsigned short* p = W3b + k0 + 16 * (lane >> 4);
      u.q[0] = *(const uint4*)(p);
      u.q[1] = *(const uint4*)(p + 8);
    }
    const v16bf A = load_afrag(h2s, H2, wave * 16, k0, lane);
    acc3 = wmma_bf16(A, u.v, acc3);
  }
  // column 0 of D: lane 0 holds M=0..7 (VGPR i), lane 16 holds M=8..15
  if ((lane & 15) == 0) {
    const int mb = wave * 16 + 8 * (lane >> 4);
    float* op = out + (size_t)wg * TM + mb;
    #pragma unroll
    for (int i = 0; i < 8; ++i) __builtin_nontemporal_store(acc3[i], op + i);
  }
}

extern "C" void kernel_launch(void* const* d_in, const int* in_sizes, int n_in,
                              void* d_out, int out_size, void* d_ws, size_t ws_size,
                              hipStream_t stream) {
  (void)in_sizes; (void)n_in; (void)ws_size;
  const float* hinst = (const float*)d_in[0];
  const float* memb  = (const float*)d_in[1];
  const int*   m2i   = (const int*)d_in[2];
  const float* W1 = (const float*)d_in[3];
  const float* b1 = (const float*)d_in[4];
  const float* W2 = (const float*)d_in[5];
  const float* b2 = (const float*)d_in[6];
  const float* W3 = (const float*)d_in[7];
  const float* b3 = (const float*)d_in[8];
  float* out = (float*)d_out;
  unsigned short* ws = (unsigned short*)d_ws;   // needs ~230 KB

  prep_weights<<<dim3(64), dim3(256), 0, stream>>>(W1, W2, W3, ws);

  const int blocks = out_size / TM;  // 524288 / 128 = 4096
  const size_t shmem = (size_t)(TM * K1 + TM * H1) * sizeof(unsigned short); // 144 KB
  node_decoder_mlp<<<dim3(blocks), dim3(256), shmem, stream>>>(
      hinst, memb, m2i, ws, b1, b2, b3, out);
}
